// AdaptiveFourier3d_59296318489249
// MI455X (gfx1250) — matrix operand order repaired and checked by reading, not compiled, not run on hardware
//
#include <hip/hip_runtime.h>
#include <hip/hip_bf16.h>
#include <math.h>

// ---------------------------------------------------------------------------
// AFNO-3D for MI455X (gfx1250, wave32, WMMA + async-to-LDS)
//   pass 1: rFFT32 along D        (f32, per-thread unrolled radix-2)
//   pass 2: FFT32 along W         (complex, in place)
//   pass 3: FFT32 along H         (complex, in place)
//   pass 4: block-diag complex MLP: global_load_async_to_lds_b128 staging,
//           v_wmma_f32_16x16x32_f16 compute, branch-free erf-GELU epilogue
//   pass 5: iFFT32 along H
//   pass 6: iFFT32 along W
//   pass 7: irFFT32 along D + residual bias -> d_out
// Spectral scratch: [B=2][C=768][H=32][W=32][Dr=17] float2 in d_ws (~204 MiB)
// ---------------------------------------------------------------------------

#define B_  2
#define C_  768
#define NB_ 8
#define BS_ 96
#define H_  32
#define W_  32
#define D_  32
#define DR_ 17
#define M_  (H_ * W_ * DR_)        // 17408 modes per channel
#define MT_ 64                     // modes per LDS tile
#define LAM_ 0.01f

typedef _Float16 v16h __attribute__((ext_vector_type(16)));
typedef float    v8f  __attribute__((ext_vector_type(8)));

// cos(pi*k/16), sin(pi*k/16) for k = 0..15
__device__ __constant__ float COS_T[16] = {
    1.0f,                0.9807852804032304f, 0.9238795325112867f, 0.8314696123025452f,
    0.7071067811865476f, 0.5555702330196022f, 0.3826834323650898f, 0.1950903220161283f,
    0.0f,               -0.1950903220161283f,-0.3826834323650898f,-0.5555702330196022f,
   -0.7071067811865476f,-0.8314696123025452f,-0.9238795325112867f,-0.9807852804032304f};
__device__ __constant__ float SIN_T[16] = {
    0.0f,                0.1950903220161283f, 0.3826834323650898f, 0.5555702330196022f,
    0.7071067811865476f, 0.8314696123025452f, 0.9238795325112867f, 0.9807852804032304f,
    1.0f,                0.9807852804032304f, 0.9238795325112867f, 0.8314696123025452f,
    0.7071067811865476f, 0.5555702330196022f, 0.3826834323650898f, 0.1950903220161283f};

#define ORTHO_SC 0.17677669529663689f   // 1/sqrt(32)

// Fully unrolled in-register 32-point radix-2 DIT FFT.  SGN=-1 fwd, +1 inv.
template <int SGN>
__device__ __forceinline__ void fft32(float re[32], float im[32]) {
#pragma unroll
  for (int i = 0; i < 32; ++i) {
    const int j = ((i & 1) << 4) | ((i & 2) << 2) | (i & 4) | ((i & 8) >> 2) | ((i & 16) >> 4);
    if (j > i) {
      float t = re[i]; re[i] = re[j]; re[j] = t;
      t = im[i]; im[i] = im[j]; im[j] = t;
    }
  }
#pragma unroll
  for (int s = 1; s <= 5; ++s) {
    const int len = 1 << s, half = len >> 1, step = 32 >> s;
#pragma unroll
    for (int i = 0; i < 32; i += len) {
#pragma unroll
      for (int k = 0; k < half; ++k) {
        const float c  = COS_T[k * step];
        const float sn = (float)SGN * SIN_T[k * step];
        const float tr = re[i + k + half], ti = im[i + k + half];
        const float wr = tr * c - ti * sn;
        const float wi = tr * sn + ti * c;
        const float ur = re[i + k], ui = im[i + k];
        re[i + k]        = ur + wr;  im[i + k]        = ui + wi;
        re[i + k + half] = ur - wr;  im[i + k + half] = ui - wi;
      }
    }
  }
}

// ---- pass 1: real FFT along D (contiguous), keep 17 bins -------------------
__global__ __launch_bounds__(256) void fft_d_kernel(const float* __restrict__ x,
                                                    float2* __restrict__ spec, int nlines) {
  const int line = blockIdx.x * blockDim.x + threadIdx.x;
  if (line >= nlines) return;
  const float* p = x + (size_t)line * D_;
  float re[32], im[32];
#pragma unroll
  for (int i = 0; i < 32; ++i) { re[i] = p[i]; im[i] = 0.0f; }
  fft32<-1>(re, im);
  float2* q = spec + (size_t)line * DR_;
#pragma unroll
  for (int i = 0; i < DR_; ++i) q[i] = make_float2(re[i] * ORTHO_SC, im[i] * ORTHO_SC);
}

// ---- passes 2/3/5/6: complex FFT along a strided axis, in place ------------
template <int SGN>
__global__ __launch_bounds__(256) void fft_pass_kernel(float2* __restrict__ spec,
                                                       int stride, int nlines) {
  const int line = blockIdx.x * blockDim.x + threadIdx.x;
  if (line >= nlines) return;
  const size_t base = (size_t)(line / stride) * ((size_t)stride * 32) + (size_t)(line % stride);
  float re[32], im[32];
#pragma unroll
  for (int i = 0; i < 32; ++i) {
    const float2 v = spec[base + (size_t)i * stride];
    re[i] = v.x; im[i] = v.y;
  }
  fft32<SGN>(re, im);
#pragma unroll
  for (int i = 0; i < 32; ++i)
    spec[base + (size_t)i * stride] = make_float2(re[i] * ORTHO_SC, im[i] * ORTHO_SC);
}

// ---- pass 7: Hermitian reconstruct + inverse real FFT along D + bias -------
__global__ __launch_bounds__(256) void ifft_d_bias_kernel(const float2* __restrict__ spec,
                                                          const float* __restrict__ x,
                                                          float* __restrict__ out, int nlines) {
  const int line = blockIdx.x * blockDim.x + threadIdx.x;
  if (line >= nlines) return;
  const float2* q = spec + (size_t)line * DR_;
  float re[32], im[32];
#pragma unroll
  for (int i = 0; i < DR_; ++i) { re[i] = q[i].x; im[i] = q[i].y; }
#pragma unroll
  for (int k = DR_; k < 32; ++k) { re[k] = q[32 - k].x; im[k] = -q[32 - k].y; }
  fft32<1>(re, im);
  const float* xb = x + (size_t)line * D_;
  float* o = out + (size_t)line * D_;
#pragma unroll
  for (int i = 0; i < 32; ++i) o[i] = re[i] * ORTHO_SC + xb[i];
}

// ---- pass 4: block-diagonal complex MLP via WMMA ---------------------------

// A-frag (16x32 f16): lane&15 = row M; lanes>=16 hold K+16..K+31.
__device__ __forceinline__ v16h load_a_frag(const _Float16* __restrict__ W, int orow,
                                            int k0, int lane) {
  const int m   = orow + (lane & 15);
  const int klo = k0 + ((lane >> 4) << 4);
  const _Float16* p = W + m * BS_;
  v16h a;
#pragma unroll
  for (int j = 0; j < 8; ++j) a[j] = p[klo + j];
#pragma unroll
  for (int j = 0; j < 8; ++j) a[8 + j] = p[klo + 8 + j];
  return a;
}

// B-frag pair from f32 complex LDS tile: one ds_load_b64 feeds both frags.
__device__ __forceinline__ void load_b_frag_c(const float2* __restrict__ X, int ncol,
                                              int k0, int lane, v16h& br, v16h& bi) {
  const int col = ncol + (lane & 15);
  const int kb  = k0 + ((lane >> 4) << 4);
#pragma unroll
  for (int j = 0; j < 16; ++j) {
    const float2 v = X[(kb + j) * MT_ + col];
    br[j] = (_Float16)v.x;
    bi[j] = (_Float16)v.y;
  }
}

// B-frag from f16 LDS (hidden activations).
__device__ __forceinline__ v16h load_b_frag_h(const _Float16* __restrict__ X, int ncol,
                                              int k0, int lane) {
  const int col = ncol + (lane & 15);
  const int kb  = k0 + ((lane >> 4) << 4);
  v16h b;
#pragma unroll
  for (int j = 0; j < 16; ++j) b[j] = X[(kb + j) * MT_ + col];
  return b;
}

// Branch-free exact-erf GELU (Abramowitz-Stegun 7.1.26, |err|<=1.5e-7):
// avoids the divergent slow path of libm erff between WMMA issues.
__device__ __forceinline__ float gelu_exact(float v) {
  const float ax = fabsf(v) * 0.70710678118654752f;
  const float t  = __builtin_amdgcn_rcpf(1.0f + 0.3275911f * ax);
  const float p  = t * (0.254829592f +
                   t * (-0.284496736f +
                   t * (1.421413741f +
                   t * (-1.453152027f +
                   t *  1.061405429f))));
  const float ea = 1.0f - p * __expf(-ax * ax);
  return 0.5f * v * (1.0f + copysignf(ea, v));
}
__device__ __forceinline__ float softshrink(float v) {
  const float a = fabsf(v) - LAM_;
  return (a > 0.0f) ? copysignf(a, v) : 0.0f;
}

__global__ __launch_bounds__(256) void mix_kernel(
    float2* __restrict__ spec,
    const float* __restrict__ w1r, const float* __restrict__ w1i,
    const float* __restrict__ w2r, const float* __restrict__ w2i,
    const float* __restrict__ b1r, const float* __restrict__ b1i,
    const float* __restrict__ b2r, const float* __restrict__ b2i) {
  __shared__ _Float16 sW1r[BS_ * BS_], sW1i[BS_ * BS_];
  __shared__ _Float16 sW2r[BS_ * BS_], sW2i[BS_ * BS_];
  __shared__ __align__(16) float2 sXf[BS_ * MT_];          // f32 complex, DMA target
  __shared__ _Float16 sHr[BS_ * MT_], sHi[BS_ * MT_];
  __shared__ float sB1r[BS_], sB1i[BS_], sB2r[BS_], sB2i[BS_];

  const int n = blockIdx.y, b = blockIdx.z;
  const int tid = threadIdx.x, lane = tid & 31, wave = tid >> 5;

  __builtin_prefetch(w1r + (size_t)n * BS_ * BS_, 0, 1);   // global_prefetch_b8
  __builtin_prefetch(w2r + (size_t)n * BS_ * BS_, 0, 1);

  // Stage weights transposed to [o][i] as f16 (needs convert -> VALU path).
  for (int idx = tid; idx < BS_ * BS_; idx += 256) {
    const int i = idx / BS_, o = idx % BS_;
    const size_t g = ((size_t)n * BS_ + i) * BS_ + o;
    sW1r[o * BS_ + i] = (_Float16)w1r[g];
    sW1i[o * BS_ + i] = (_Float16)w1i[g];
    sW2r[o * BS_ + i] = (_Float16)w2r[g];
    sW2i[o * BS_ + i] = (_Float16)w2i[g];
  }
  for (int idx = tid; idx < BS_; idx += 256) {
    sB1r[idx] = b1r[n * BS_ + idx]; sB1i[idx] = b1i[n * BS_ + idx];
    sB2r[idx] = b2r[n * BS_ + idx]; sB2i[idx] = b2i[n * BS_ + idx];
  }

  float2* Xg = spec + ((size_t)(b * C_ + n * BS_)) * M_;   // [96 ch][M_] complex

  for (int t = 0; t < 4; ++t) {
    const int m0 = (blockIdx.x * 4 + t) * MT_;

    // ---- async DMA the activation tile into LDS (ASYNCcnt path) ----
    // 16B per lane per issue: 96 rows x 32 chunks of 2 complex f32.
    for (int idx = tid; idx < BS_ * 32; idx += 256) {
      const int i = idx >> 5, mc = idx & 31;
      const float2* g = Xg + (size_t)i * M_ + (size_t)(m0 + mc * 2);
      const float2* l = sXf + i * MT_ + mc * 2;
      const unsigned lds_lo = (unsigned)(unsigned long long)l;   // flat lo32 == LDS addr
      asm volatile("global_load_async_to_lds_b128 %0, %1, off"
                   :: "v"(lds_lo), "v"(g) : "memory");
    }
    asm volatile("s_wait_asynccnt 0x0" ::: "memory");
    __syncthreads();   // weights + X visible; prior-iter GEMM2 drained

    // ---- layer 1: h1 = gelu(W1^T x + b1), complex via 4 real WMMAs ----
    for (int tile = wave; tile < 24; tile += 8) {
      const int orow = (tile / 4) * 16, ncol = (tile % 4) * 16;
      v8f arr = {0,0,0,0,0,0,0,0}, aii = {0,0,0,0,0,0,0,0};
      v8f ari = {0,0,0,0,0,0,0,0}, air = {0,0,0,0,0,0,0,0};
#pragma unroll
      for (int kc = 0; kc < 3; ++kc) {
        const int k0 = kc * 32;
        const v16h awr = load_a_frag(sW1r, orow, k0, lane);
        const v16h awi = load_a_frag(sW1i, orow, k0, lane);
        v16h bxr, bxi;
        load_b_frag_c(sXf, ncol, k0, lane, bxr, bxi);
        arr = __builtin_amdgcn_wmma_f32_16x16x32_f16(false, awr, false, bxr, (short)0, arr, false, false);
        aii = __builtin_amdgcn_wmma_f32_16x16x32_f16(false, awi, false, bxi, (short)0, aii, false, false);
        ari = __builtin_amdgcn_wmma_f32_16x16x32_f16(false, awr, false, bxi, (short)0, ari, false, false);
        air = __builtin_amdgcn_wmma_f32_16x16x32_f16(false, awi, false, bxr, (short)0, air, false, false);
      }
      const int rbase = orow + ((lane >> 4) << 3);
      const int col = ncol + (lane & 15);
#pragma unroll
      for (int v = 0; v < 8; ++v) {
        const int row = rbase + v;
        const float re = arr[v] - aii[v] + sB1r[row];
        const float im = ari[v] + air[v] + sB1i[row];
        sHr[row * MT_ + col] = (_Float16)gelu_exact(re);
        sHi[row * MT_ + col] = (_Float16)gelu_exact(im);
      }
    }
    __syncthreads();

    // ---- layer 2: h2 = softshrink(W2^T h1 + b2), write back in place ----
    for (int tile = wave; tile < 24; tile += 8) {
      const int orow = (tile / 4) * 16, ncol = (tile % 4) * 16;
      v8f arr = {0,0,0,0,0,0,0,0}, aii = {0,0,0,0,0,0,0,0};
      v8f ari = {0,0,0,0,0,0,0,0}, air = {0,0,0,0,0,0,0,0};
#pragma unroll
      for (int kc = 0; kc < 3; ++kc) {
        const int k0 = kc * 32;
        const v16h awr = load_a_frag(sW2r, orow, k0, lane);
        const v16h awi = load_a_frag(sW2i, orow, k0, lane);
        const v16h bhr = load_b_frag_h(sHr, ncol, k0, lane);
        const v16h bhi = load_b_frag_h(sHi, ncol, k0, lane);
        arr = __builtin_amdgcn_wmma_f32_16x16x32_f16(false, awr, false, bhr, (short)0, arr, false, false);
        aii = __builtin_amdgcn_wmma_f32_16x16x32_f16(false, awi, false, bhi, (short)0, aii, false, false);
        ari = __builtin_amdgcn_wmma_f32_16x16x32_f16(false, awr, false, bhi, (short)0, ari, false, false);
        air = __builtin_amdgcn_wmma_f32_16x16x32_f16(false, awi, false, bhr, (short)0, air, false, false);
      }
      const int rbase = orow + ((lane >> 4) << 3);
      const int col = ncol + (lane & 15);
#pragma unroll
      for (int v = 0; v < 8; ++v) {
        const int row = rbase + v;
        const float re = softshrink(arr[v] - aii[v] + sB2r[row]);
        const float im = softshrink(ari[v] + air[v] + sB2i[row]);
        Xg[(size_t)row * M_ + (size_t)(m0 + col)] = make_float2(re, im);
      }
    }
    __syncthreads();   // sH/sXf reads done before next iter's DMA overwrites
  }
}

// ---------------------------------------------------------------------------
extern "C" void kernel_launch(void* const* d_in, const int* in_sizes, int n_in,
                              void* d_out, int out_size, void* d_ws, size_t ws_size,
                              hipStream_t stream) {
  (void)in_sizes; (void)n_in; (void)out_size; (void)ws_size;
  const float* x   = (const float*)d_in[0];
  const float* w1r = (const float*)d_in[1];
  const float* w1i = (const float*)d_in[2];
  const float* w2r = (const float*)d_in[3];
  const float* w2i = (const float*)d_in[4];
  const float* b1r = (const float*)d_in[5];
  const float* b1i = (const float*)d_in[6];
  const float* b2r = (const float*)d_in[7];
  const float* b2i = (const float*)d_in[8];
  float* out = (float*)d_out;
  float2* spec = (float2*)d_ws;    // [B][C][H][W][DR_] complex, ~204 MiB

  const int NL_D  = B_ * C_ * H_ * W_;        // 1,572,864 D-lines
  const int NL_WH = B_ * C_ * H_ * DR_;       //   835,584 W/H-lines

  fft_d_kernel<<<(NL_D + 255) / 256, 256, 0, stream>>>(x, spec, NL_D);
  fft_pass_kernel<-1><<<(NL_WH + 255) / 256, 256, 0, stream>>>(spec, DR_, NL_WH);        // W
  fft_pass_kernel<-1><<<(NL_WH + 255) / 256, 256, 0, stream>>>(spec, W_ * DR_, NL_WH);   // H

  dim3 mixGrid(M_ / (MT_ * 4), NB_, B_);      // (68, 8, 2)
  mix_kernel<<<mixGrid, 256, 0, stream>>>(spec, w1r, w1i, w2r, w2i, b1r, b1i, b2r, b2i);

  fft_pass_kernel<1><<<(NL_WH + 255) / 256, 256, 0, stream>>>(spec, W_ * DR_, NL_WH);    // iH
  fft_pass_kernel<1><<<(NL_WH + 255) / 256, 256, 0, stream>>>(spec, DR_, NL_WH);         // iW
  ifft_d_bias_kernel<<<(NL_D + 255) / 256, 256, 0, stream>>>(spec, x, out, NL_D);
}
